// fusionWithCrossAttention_75411035783633
// MI455X (gfx1250) — matrix operand order repaired and checked by reading, not compiled, not run on hardware
//
#include <hip/hip_runtime.h>
#include <hip/hip_fp16.h>

// Flip to 0 if the async-to-LDS inline asm fails to assemble on this toolchain.
#ifndef USE_ASYNC_COPY
#define USE_ASYNC_COPY 1
#endif

// ---------- types ----------
typedef _Float16 h16;
typedef __attribute__((ext_vector_type(16))) _Float16 v16h;
typedef __attribute__((ext_vector_type(8)))  _Float16 v8h;
typedef __attribute__((ext_vector_type(4)))  _Float16 v4h;
typedef __attribute__((ext_vector_type(8)))  float    v8f;

#define BN_EPS 1e-5f

// Generic LDS pointer -> wave-relative LDS byte offset.
// ISA 10.2 aperture mapping: for LDS-space flat addresses, the LDS address is the low 32 bits.
__device__ __forceinline__ unsigned lds_off(const void* p) {
    return (unsigned)(unsigned long long)p;
}

// ---------- f32 -> f16 conversion (vectorized x4) ----------
__global__ __launch_bounds__(256)
void cvt_f32_f16(const float* __restrict__ in, h16* __restrict__ out, int n4) {
    int i = blockIdx.x * blockDim.x + threadIdx.x;
    if (i < n4) {
        float4 f = ((const float4*)in)[i];
        v4h o;
        o[0] = (h16)f.x; o[1] = (h16)f.y; o[2] = (h16)f.z; o[3] = (h16)f.w;
        ((v4h*)out)[i] = o;
    }
}

// ---------- WMMA GEMM: C = epilogue(A[M,K] @ W[N,K]^T + bias) ----------
// Block: 256 threads (8 waves). Block tile 128(M) x 64(N), K-step 32.
// Wave tile 32x32 = 2x2 v_wmma_f32_16x16x32_f16 accumulators.
// Double-buffered LDS filled by global_load_async_to_lds_b128 (ASYNCcnt),
// overlapping the k+1 tile DMA with the k-step WMMAs.
enum { EP_BN_RELU = 0, EP_BIAS_H = 1, EP_BIAS_F = 2 };

template<int MODE>
__global__ __launch_bounds__(256)
void gemm_wmma(const h16* __restrict__ A, int lda,
               const h16* __restrict__ W,                 // [N x K], row-major
               const float* __restrict__ bias,
               const float* __restrict__ g,  const float* __restrict__ bb,
               const float* __restrict__ mn, const float* __restrict__ vr,
               void* __restrict__ Cp, long ldc, long coff, int K)
{
    constexpr int BK = 32, LDSS = 40;                     // 80B padded LDS row
    __shared__ h16 As[2][128 * LDSS];                     // 2 x 10.0 KB
    __shared__ h16 Bs[2][64 * LDSS];                      // 2 x  5.0 KB

    const int t     = threadIdx.x;
    const int lane  = t & 31;
    const int w     = t >> 5;
    const int wm    = w & 3;                              // 4 waves along M
    const int wn    = w >> 2;                             // 2 waves along N
    const int lhalf = lane & 15;
    const int hig   = lane >> 4;                          // lane group 0/1

    const long tileM = (long)blockIdx.x * 128;
    const long tileN = (long)blockIdx.y * 64;

    // cooperative tile DMA: A 128x32 halves (16/thr = 2 x b128), B 64x32 (8/thr = 1 x b128)
    const int arow = t >> 1, acol = (t & 1) * 16;
    const int brow = t >> 2, bcol = (t & 3) * 8;
    const h16* aSrc = A + (tileM + arow) * (long)lda + acol;
    const h16* bSrc = W + (tileN + brow) * (long)K + bcol;
    unsigned aL[2] = { lds_off(&As[0][arow * LDSS + acol]), lds_off(&As[1][arow * LDSS + acol]) };
    unsigned bL[2] = { lds_off(&Bs[0][brow * LDSS + bcol]), lds_off(&Bs[1][brow * LDSS + bcol]) };

    v8f acc[2][2] = {};

    const int nk = K / BK;

#if USE_ASYNC_COPY
    // Issue helper: 3 async b128 DMAs per thread per tile (same INST_OFFSET is
    // applied to both the global and LDS addresses, ISA 10 async pseudocode).
    #define ISSUE_TILE(k0, buf)                                                          \
        do {                                                                             \
            const h16* ap_ = aSrc + (k0);                                                \
            const h16* bp_ = bSrc + (k0);                                                \
            asm volatile("global_load_async_to_lds_b128 %0, %1, off"                     \
                         :: "v"(aL[buf]), "v"(ap_) : "memory");                          \
            asm volatile("global_load_async_to_lds_b128 %0, %1, off offset:16"          \
                         :: "v"(aL[buf]), "v"(ap_) : "memory");                          \
            asm volatile("global_load_async_to_lds_b128 %0, %1, off"                     \
                         :: "v"(bL[buf]), "v"(bp_) : "memory");                          \
        } while (0)

    ISSUE_TILE(0, 0);
#endif

    for (int i = 0; i < nk; ++i) {
        const int buf = i & 1;

#if USE_ASYNC_COPY
        if (i + 1 < nk) {
            ISSUE_TILE((i + 1) * BK, buf ^ 1);            // prefetch next tile into other buffer
            asm volatile("s_wait_asynccnt 0x3" ::: "memory");  // tile i done; tile i+1 in flight
        } else {
            asm volatile("s_wait_asynccnt 0x0" ::: "memory");
        }
        __syncthreads();                                  // tile i visible to all waves
#else
        __syncthreads();
        {
            const int k0 = i * BK;
            v8h a0 = *(const v8h*)(aSrc + k0);
            v8h a1 = *(const v8h*)(aSrc + k0 + 8);
            *(v8h*)&As[buf][arow * LDSS + acol]     = a0;
            *(v8h*)&As[buf][arow * LDSS + acol + 8] = a1;
            *(v8h*)&Bs[buf][brow * LDSS + bcol] = *(const v8h*)(bSrc + k0);
            if (k0 + BK < K) {
                __builtin_prefetch(aSrc + k0 + BK, 0, 1);
                __builtin_prefetch(bSrc + k0 + BK, 0, 1);
            }
        }
        __syncthreads();
#endif

        // ISA 7.12.2 fragment layouts (wave32, 16-bit):
        // A 16x32: lane<16 holds K{0..7,16..23}, lane>=16 holds K{8..15,24..31}
        // B 32x16: lane<16 holds K{0..7,16..23}, lane>=16 holds K{8..15,24..31} (N = lane%16)
        v16h af[2], bf[2];
#pragma unroll
        for (int mt = 0; mt < 2; ++mt) {
            int r  = wm * 32 + mt * 16 + lhalf;
            int ka = hig * 8;
            v8h lo = *(const v8h*)&As[buf][r * LDSS + ka];
            v8h hi = *(const v8h*)&As[buf][r * LDSS + ka + 16];
            af[mt] = __builtin_shufflevector(lo, hi, 0,1,2,3,4,5,6,7,8,9,10,11,12,13,14,15);
        }
#pragma unroll
        for (int nt = 0; nt < 2; ++nt) {
            int c  = wn * 32 + nt * 16 + lhalf;
            int kb = hig * 8;
            v8h lo = *(const v8h*)&Bs[buf][c * LDSS + kb];
            v8h hi = *(const v8h*)&Bs[buf][c * LDSS + kb + 16];
            bf[nt] = __builtin_shufflevector(lo, hi, 0,1,2,3,4,5,6,7,8,9,10,11,12,13,14,15);
        }
#pragma unroll
        for (int mt = 0; mt < 2; ++mt)
#pragma unroll
            for (int nt = 0; nt < 2; ++nt)
                acc[mt][nt] = __builtin_amdgcn_wmma_f32_16x16x32_f16(
                    false, af[mt], false, bf[nt], (short)0, acc[mt][nt], false, false);

        __syncthreads();                                  // reads done before buf is re-filled
    }

    // epilogue: C 16x16 f32 frag = 8 VGPRs; M = r + 8*lanegroup, N = lane%16
#pragma unroll
    for (int nt = 0; nt < 2; ++nt) {
        long col = tileN + wn * 32 + nt * 16 + lhalf;
        float bv = bias[col];
        float scale = 1.f, shift = 0.f;
        if (MODE == EP_BN_RELU) {
            float s = g[col] * rsqrtf(vr[col] + BN_EPS);
            scale = s;
            shift = bb[col] - mn[col] * s;
        }
#pragma unroll
        for (int mt = 0; mt < 2; ++mt) {
            long row0 = tileM + wm * 32 + mt * 16 + hig * 8;
#pragma unroll
            for (int r = 0; r < 8; ++r) {
                float x = acc[mt][nt][r] + bv;
                if (MODE == EP_BN_RELU) { x = x * scale + shift; x = x > 0.f ? x : 0.f; }
                long idx = coff + (row0 + r) * ldc + col;
                if (MODE == EP_BIAS_F) ((float*)Cp)[idx] = x;
                else                   ((h16*)Cp)[idx]   = (h16)x;
            }
        }
    }
}

// ---------- attention: seq-len 2 per (b,h); one wave32 per (b,h) ----------
// q/k/v layout: row (b*2+s), col (h*64+d).  lane covers d = 2*lane, 2*lane+1.
__global__ __launch_bounds__(256)
void attn_kernel(const h16* __restrict__ q, const h16* __restrict__ k,
                 const h16* __restrict__ v, h16* __restrict__ o)
{
    int pair = blockIdx.x * 8 + (threadIdx.x >> 5);   // b*H + h
    int lane = threadIdx.x & 31;
    int b = pair >> 3, h = pair & 7;                  // H = 8
    long base0 = ((long)(b * 2 + 0)) * 512 + h * 64 + lane * 2;
    long base1 = ((long)(b * 2 + 1)) * 512 + h * 64 + lane * 2;

    float q0a = (float)q[base0], q0b = (float)q[base0 + 1];
    float q1a = (float)q[base1], q1b = (float)q[base1 + 1];
    float k0a = (float)k[base0], k0b = (float)k[base0 + 1];
    float k1a = (float)k[base1], k1b = (float)k[base1 + 1];

    float s00 = q0a * k0a + q0b * k0b;
    float s01 = q0a * k1a + q0b * k1b;
    float s10 = q1a * k0a + q1b * k0b;
    float s11 = q1a * k1a + q1b * k1b;
#pragma unroll
    for (int off = 16; off; off >>= 1) {
        s00 += __shfl_xor(s00, off, 32);
        s01 += __shfl_xor(s01, off, 32);
        s10 += __shfl_xor(s10, off, 32);
        s11 += __shfl_xor(s11, off, 32);
    }
    const float sc = 0.125f;                          // 1/sqrt(64)
    s00 *= sc; s01 *= sc; s10 *= sc; s11 *= sc;
    float m0 = fmaxf(s00, s01), m1 = fmaxf(s10, s11);
    float e00 = __expf(s00 - m0), e01 = __expf(s01 - m0);
    float e10 = __expf(s10 - m1), e11 = __expf(s11 - m1);
    float i0 = 1.f / (e00 + e01), i1 = 1.f / (e10 + e11);
    float a00 = e00 * i0, a01 = e01 * i0;
    float a10 = e10 * i1, a11 = e11 * i1;

    float v0a = (float)v[base0], v0b = (float)v[base0 + 1];
    float v1a = (float)v[base1], v1b = (float)v[base1 + 1];
    o[base0]     = (h16)(a00 * v0a + a01 * v1a);
    o[base0 + 1] = (h16)(a00 * v0b + a01 * v1b);
    o[base1]     = (h16)(a10 * v0a + a11 * v1a);
    o[base1 + 1] = (h16)(a10 * v0b + a11 * v1b);
}

// ---------- final head: [B,1024] @ [3,1024]^T + b ----------
__global__ __launch_bounds__(96)
void head_kernel(const float* __restrict__ fused, const float* __restrict__ w,
                 const float* __restrict__ bias, float* __restrict__ out)
{
    int b = blockIdx.x;
    int j = threadIdx.x >> 5;                         // 3 waves, one output each
    int lane = threadIdx.x & 31;
    const float* row = fused + (long)b * 1024;
    const float* wr  = w + j * 1024;
    float s = 0.f;
#pragma unroll 4
    for (int i = lane; i < 1024; i += 32) s += row[i] * wr[i];
#pragma unroll
    for (int off = 16; off; off >>= 1) s += __shfl_xor(s, off, 32);
    if (lane == 0) out[b * 3 + j] = s + bias[j];
}

// ---------- launcher ----------
extern "C" void kernel_launch(void* const* d_in, const int* in_sizes, int n_in,
                              void* d_out, int out_size, void* d_ws, size_t ws_size,
                              hipStream_t stream) {
    const int B = 16384, VDIM = 2048, TDIM = 768, D = 512;

    const float* v_emb  = (const float*)d_in[0];
    const float* t_emb  = (const float*)d_in[1];
    const float* vfc_w  = (const float*)d_in[2];
    const float* vfc_b  = (const float*)d_in[3];
    const float* tfc_w  = (const float*)d_in[4];
    const float* tfc_b  = (const float*)d_in[5];
    const float* bn1_g  = (const float*)d_in[6];
    const float* bn1_b  = (const float*)d_in[7];
    const float* bn1_m  = (const float*)d_in[8];
    const float* bn1_v  = (const float*)d_in[9];
    const float* bn2_g  = (const float*)d_in[10];
    const float* bn2_b  = (const float*)d_in[11];
    const float* bn2_m  = (const float*)d_in[12];
    const float* bn2_v  = (const float*)d_in[13];
    const float* tq_w = (const float*)d_in[14]; const float* tq_b = (const float*)d_in[15];
    const float* tk_w = (const float*)d_in[16]; const float* tk_b = (const float*)d_in[17];
    const float* tv_w = (const float*)d_in[18]; const float* tv_b = (const float*)d_in[19];
    const float* vq_w = (const float*)d_in[20]; const float* vq_b = (const float*)d_in[21];
    const float* vk_w = (const float*)d_in[22]; const float* vk_b = (const float*)d_in[23];
    const float* vv_w = (const float*)d_in[24]; const float* vv_b = (const float*)d_in[25];
    const float* in_proj_w  = (const float*)d_in[26];
    const float* in_proj_b  = (const float*)d_in[27];
    const float* out_proj_w = (const float*)d_in[28];
    const float* out_proj_b = (const float*)d_in[29];
    const float* output_w   = (const float*)d_in[30];
    const float* output_b   = (const float*)d_in[31];

    // workspace carve-up (256B aligned)
    char* ws = (char*)d_ws;
    size_t off = 0;
    auto take = [&](size_t elems, size_t esz) -> void* {
        void* p = ws + off;
        off = (off + elems * esz + 255) & ~(size_t)255;
        return p;
    };
    h16* hve   = (h16*)take((size_t)B * VDIM, 2);
    h16* hte   = (h16*)take((size_t)B * TDIM, 2);
    h16* hvfc  = (h16*)take((size_t)D * VDIM, 2);
    h16* htfc  = (h16*)take((size_t)D * TDIM, 2);
    h16* htq   = (h16*)take((size_t)D * D, 2);
    h16* htk   = (h16*)take((size_t)D * D, 2);
    h16* htv   = (h16*)take((size_t)D * D, 2);
    h16* hvq   = (h16*)take((size_t)D * D, 2);
    h16* hvk   = (h16*)take((size_t)D * D, 2);
    h16* hvv   = (h16*)take((size_t)D * D, 2);
    h16* hinp  = (h16*)take((size_t)3 * D * D, 2);
    h16* houtp = (h16*)take((size_t)D * D, 2);
    h16* hv    = (h16*)take((size_t)B * D, 2);       // relu(bn(vfc(v_emb)))
    h16* ht    = (h16*)take((size_t)B * D, 2);
    h16* hquery= (h16*)take((size_t)2 * B * D, 2);
    h16* hkey  = (h16*)take((size_t)2 * B * D, 2);
    h16* hval  = (h16*)take((size_t)2 * B * D, 2);
    h16* hq    = (h16*)take((size_t)2 * B * D, 2);
    h16* hk    = (h16*)take((size_t)2 * B * D, 2);
    h16* hvv2  = (h16*)take((size_t)2 * B * D, 2);
    h16* ho    = (h16*)take((size_t)2 * B * D, 2);
    float* o2f = (float*)take((size_t)2 * B * D, 4);

    auto cvt = [&](const float* src, h16* dst, size_t n) {
        int n4 = (int)(n / 4);
        cvt_f32_f16<<<(n4 + 255) / 256, 256, 0, stream>>>(src, dst, n4);
    };
    cvt(v_emb, hve, (size_t)B * VDIM);
    cvt(t_emb, hte, (size_t)B * TDIM);
    cvt(vfc_w, hvfc, (size_t)D * VDIM);
    cvt(tfc_w, htfc, (size_t)D * TDIM);
    cvt(tq_w, htq, (size_t)D * D);  cvt(tk_w, htk, (size_t)D * D);  cvt(tv_w, htv, (size_t)D * D);
    cvt(vq_w, hvq, (size_t)D * D);  cvt(vk_w, hvk, (size_t)D * D);  cvt(vv_w, hvv, (size_t)D * D);
    cvt(in_proj_w, hinp, (size_t)3 * D * D);
    cvt(out_proj_w, houtp, (size_t)D * D);

    const dim3 blk(256);
    const dim3 g1(B / 128, D / 64);        // M = B
    const dim3 g2(2 * B / 128, D / 64);    // M = 2B

    // stage 1: v / t with fused BN + ReLU
    gemm_wmma<EP_BN_RELU><<<g1, blk, 0, stream>>>(hve, VDIM, hvfc, vfc_b,
        bn1_g, bn1_b, bn1_m, bn1_v, hv, D, 0, VDIM);
    gemm_wmma<EP_BN_RELU><<<g1, blk, 0, stream>>>(hte, TDIM, htfc, tfc_b,
        bn2_g, bn2_b, bn2_m, bn2_v, ht, D, 0, TDIM);

    // stage 2: query/key/value = stack(t-proj, v-proj) -> interleaved [2B, D]
    gemm_wmma<EP_BIAS_H><<<g1, blk, 0, stream>>>(ht, D, htq, tq_b, 0,0,0,0, hquery, 2 * D, 0, D);
    gemm_wmma<EP_BIAS_H><<<g1, blk, 0, stream>>>(hv, D, hvq, vq_b, 0,0,0,0, hquery, 2 * D, D, D);
    gemm_wmma<EP_BIAS_H><<<g1, blk, 0, stream>>>(ht, D, htk, tk_b, 0,0,0,0, hkey,   2 * D, 0, D);
    gemm_wmma<EP_BIAS_H><<<g1, blk, 0, stream>>>(hv, D, hvk, vk_b, 0,0,0,0, hkey,   2 * D, D, D);
    gemm_wmma<EP_BIAS_H><<<g1, blk, 0, stream>>>(ht, D, htv, tv_b, 0,0,0,0, hval,   2 * D, 0, D);
    gemm_wmma<EP_BIAS_H><<<g1, blk, 0, stream>>>(hv, D, hvv, vv_b, 0,0,0,0, hval,   2 * D, D, D);

    // stage 3: in_proj (wq | wk | wv) on [2B, D]
    gemm_wmma<EP_BIAS_H><<<g2, blk, 0, stream>>>(hquery, D, hinp,                   in_proj_b,         0,0,0,0, hq,   D, 0, D);
    gemm_wmma<EP_BIAS_H><<<g2, blk, 0, stream>>>(hkey,   D, hinp + (size_t)D * D,     in_proj_b + D,     0,0,0,0, hk,   D, 0, D);
    gemm_wmma<EP_BIAS_H><<<g2, blk, 0, stream>>>(hval,   D, hinp + (size_t)2 * D * D, in_proj_b + 2 * D, 0,0,0,0, hvv2, D, 0, D);

    // stage 4: tiny attention (seq len 2), one wave per (b,h)
    attn_kernel<<<B * 8 / 8, 256, 0, stream>>>(hq, hk, hvv2, ho);

    // stage 5: out_proj -> f32 "fused" buffer [B, 1024]
    gemm_wmma<EP_BIAS_F><<<g2, blk, 0, stream>>>(ho, D, houtp, out_proj_b, 0,0,0,0, o2f, D, 0, D);

    // stage 6: classifier head -> d_out [B,3] f32
    head_kernel<<<B, 96, 0, stream>>>(o2f, output_w, output_b, (float*)d_out);
}
